// QuestionAwareContextLayer_910533067600
// MI455X (gfx1250) — compile-verified, matrix operand order checked by reading, not run on hardware
//
#include <hip/hip_runtime.h>
#include <hip/hip_bf16.h>

// ---------------------------------------------------------------------------
// Problem constants (from the reference)
// ---------------------------------------------------------------------------
#define BN_  8
#define LC   1024
#define DC   768
#define QN   64
#define LQ   48
#define DQ   768
#define HH   1536   // 2*Dc == 2*Dq == H
#define KPAD 64     // LQ padded to WMMA K multiple

typedef __attribute__((ext_vector_type(16))) __bf16 v16bf;
typedef __attribute__((ext_vector_type(4)))  __bf16 v4bf;
typedef __attribute__((ext_vector_type(8)))  float  v8f;
typedef __attribute__((ext_vector_type(4)))  float  v4f;
typedef __attribute__((ext_vector_type(4)))  unsigned int u32x4;

__device__ __forceinline__ v8f wmma_bf16(v16bf a, v16bf b, v8f c) {
    // D(f32 16x16) = A(bf16 16x32) * B(bf16 32x16) + C
    return __builtin_amdgcn_wmma_f32_16x16x32_bf16(
        /*neg_a=*/false, a, /*neg_b=*/false, b,
        /*c_mod=*/(short)0, c, /*reuse_a=*/false, /*reuse_b=*/false);
}

// 32 contiguous bytes -> one bf16 fragment (two 128-bit loads, no scalar packing)
__device__ __forceinline__ v16bf load16v(const __bf16* p) {
    union { u32x4 u[2]; v16bf f; } t;
    const u32x4* q = (const u32x4*)p;
    t.u[0] = q[0];
    t.u[1] = q[1];
    return t.f;
}

// ---------------------------------------------------------------------------
// fp32 -> bf16 conversion, 4 elements/thread (n % 4 == 0 for all uses)
// ---------------------------------------------------------------------------
__global__ void f32_to_bf16_v4(const float* __restrict__ in, __bf16* __restrict__ out, int n4) {
    int i = blockIdx.x * blockDim.x + threadIdx.x;
    if (i >= n4) return;
    v4f x = *(const v4f*)(in + (size_t)i * 4);
    v4bf y;
    y[0] = (__bf16)x[0]; y[1] = (__bf16)x[1];
    y[2] = (__bf16)x[2]; y[3] = (__bf16)x[3];
    *(v4bf*)(out + (size_t)i * 4) = y;
}

// ---------------------------------------------------------------------------
// Question flow: per-(l,d) scan over Q with tag-group resets.
// Emits:
//   flow   [QN][LQ][HH]      row-major bf16 (A operand of q_rep GEMM)
//   flow_t [QN][HH][KPAD]    transposed + K-padded bf16 (B operand of out GEMM)
// ---------------------------------------------------------------------------
__global__ void flow_kernel(const float* __restrict__ questions,  // [QN][LQ][DQ]
                            const int*  __restrict__ tags,        // [QN]
                            __bf16*     __restrict__ flow,
                            __bf16*     __restrict__ flow_t)
{
    int idx = blockIdx.x * blockDim.x + threadIdx.x;   // over LQ*DQ
    if (idx >= LQ * DQ) return;
    int l = idx / DQ, d = idx % DQ;

    float run = 0.f;
    int pos = 0;
    int prev = tags[0];
    for (int q = 0; q < QN; ++q) {
        int tg = tags[q];
        if (tg != prev) { run = 0.f; pos = 0; prev = tg; }
        float x = questions[((size_t)q * LQ + l) * DQ + d];
        float h = (pos == 0) ? x : run / (float)pos;
        size_t base = ((size_t)q * LQ + l) * (size_t)HH;
        flow[base + d]      = (__bf16)x;
        flow[base + DQ + d] = (__bf16)h;
        size_t tb = (size_t)q * HH * KPAD;
        flow_t[tb + (size_t)d        * KPAD + l] = (__bf16)x;
        flow_t[tb + (size_t)(DQ + d) * KPAD + l] = (__bf16)h;
        run += x;
        ++pos;
    }
    // zero the K padding rows (k = 48..63) once per (d) column
    if (l == 0) {
        for (int q = 0; q < QN; ++q) {
            size_t tb = (size_t)q * HH * KPAD;
            for (int k = LQ; k < KPAD; ++k) {
                flow_t[tb + (size_t)d        * KPAD + k] = (__bf16)0.f;
                flow_t[tb + (size_t)(DQ + d) * KPAD + k] = (__bf16)0.f;
            }
        }
    }
}

// ---------------------------------------------------------------------------
// LDS-tiled WMMA GEMM:  D = act( A[M,K] @ B[K,N] ),  act = relu (* vscale[n])
// Compile-time dims -> strength-reduced addressing.
// Block tile 128x64, BK=32, 256 threads = 8 wave32s in a 4(m) x 2(n) grid;
// each wave owns a 2x2 arrangement of 16x16 tiles (4 WMMAs, A/B frag reuse).
// Double-buffered LDS: global loads for tile k+1 overlap WMMAs for tile k.
// M % 128 == 0, N % 64 == 0, K % 64 == 0 (true for all uses here).
// ---------------------------------------------------------------------------
template <int M, int N, int K>
__global__ __launch_bounds__(256) void gemm_relu_bf16(
    const __bf16* __restrict__ A, const __bf16* __restrict__ B,
    __bf16* __restrict__ D, const float* __restrict__ vscale)
{
    __shared__ __bf16 As [2][128][40];   // As[buf][m][k]   (row 80B, 16B-aligned)
    __shared__ __bf16 Bst[2][64][40];    // Bst[buf][n][k]  (transposed tile)

    const int m0 = blockIdx.x * 128;
    const int n0 = blockIdx.y * 64;
    const int t  = threadIdx.x;
    const int wave = t >> 5, lane = t & 31;
    const int wm = wave & 3, wn = wave >> 2;    // 4 x 2 wave grid
    const int m  = lane & 15, hf = lane >> 4;   // hf selects K half 0..15 / 16..31

    // cooperative-load coordinates (fixed per thread)
    const int ar = t >> 1, ac = (t & 1) * 16;   // A: 32B (two b128) per thread
    const int k2 = (t & 15) * 2;                // B: 2 k-rows x 4 cols sub-block
    const int c4 = (t >> 4) * 4;

    // staging registers (global -> reg -> LDS pipeline)
    u32x4 aS0, aS1;
    v4bf  bS0, bS1;

    auto gload = [&](int kb) {
        const __bf16* ap = A + (size_t)(m0 + ar) * K + kb + ac;
        aS0 = ((const u32x4*)ap)[0];
        aS1 = ((const u32x4*)ap)[1];
        const __bf16* bp = B + (size_t)(kb + k2) * N + n0 + c4;
        bS0 = *(const v4bf*)bp;
        bS1 = *(const v4bf*)(bp + N);
        if (kb + 32 < K) __builtin_prefetch(ap + 32, 0, 1);
    };
    auto lstore = [&](int buf) {
        *(u32x4*)&As[buf][ar][ac]     = aS0;
        *(u32x4*)&As[buf][ar][ac + 8] = aS1;
#pragma unroll
        for (int c = 0; c < 4; ++c) {
            union { __bf16 h[2]; unsigned int u; } pk;
            pk.h[0] = bS0[c];
            pk.h[1] = bS1[c];
            *(unsigned int*)&Bst[buf][c4 + c][k2] = pk.u;
        }
    };

    v8f c00 = {}, c01 = {}, c10 = {}, c11 = {};
    auto compute = [&](int buf) {
        v16bf a0 = load16v(&As [buf][wm * 32 + m][hf * 16]);
        v16bf a1 = load16v(&As [buf][wm * 32 + 16 + m][hf * 16]);
        v16bf b0 = load16v(&Bst[buf][wn * 32 + m][hf * 16]);
        v16bf b1 = load16v(&Bst[buf][wn * 32 + 16 + m][hf * 16]);
        c00 = wmma_bf16(a0, b0, c00);
        c01 = wmma_bf16(a0, b1, c01);
        c10 = wmma_bf16(a1, b0, c10);
        c11 = wmma_bf16(a1, b1, c11);
    };

    constexpr int NK = K / 32;          // even for all uses (24 or 48)
    gload(0);
    lstore(0);
    __syncthreads();

    for (int ki = 0; ki < NK; ki += 2) {
        // step A: compute buf 0, stage tile ki+1 into buf 1 (always exists)
        gload((ki + 1) * 32);
        compute(0);
        lstore(1);
        __syncthreads();
        // step B: compute buf 1, stage tile ki+2 into buf 0 (if any)
        if (ki + 2 < NK) {
            gload((ki + 2) * 32);
            compute(1);
            lstore(0);
        } else {
            compute(1);
        }
        __syncthreads();
    }

    // Epilogue: relu (+ optional column scale), store bf16.
    // C/D layout: lane L: col = L%16, VGPR r -> row = r + 8*(L/16)
    const int col0 = n0 + wn * 32 + m;
    const int col1 = col0 + 16;
    const float s0 = vscale ? vscale[col0] : 1.f;
    const float s1 = vscale ? vscale[col1] : 1.f;
#pragma unroll
    for (int r = 0; r < 8; ++r) {
        int row = m0 + wm * 32 + r + 8 * hf;
        D[(size_t)row * N + col0] = (__bf16)(fmaxf(c00[r], 0.f) * s0);
        D[(size_t)row * N + col1] = (__bf16)(fmaxf(c01[r], 0.f) * s1);
        int rowB = row + 16;
        D[(size_t)rowB * N + col0] = (__bf16)(fmaxf(c10[r], 0.f) * s0);
        D[(size_t)rowB * N + col1] = (__bf16)(fmaxf(c11[r], 0.f) * s1);
    }
}

// ---------------------------------------------------------------------------
// Fused attention per question:
//   scores = c_rep[tags[q]] @ q_rep[q]^T   (WMMA bf16, f32 acc)
//   softmax over 48                        (LDS + VALU)
//   out    = attn @ flow[q]                (WMMA bf16, f32 out, via flow_t)
// 128 threads = 4 wave32s; each wave owns 16 context rows.
// ---------------------------------------------------------------------------
__global__ __launch_bounds__(128) void attn_kernel(
    const __bf16* __restrict__ crep,    // [BN_][LC][HH]
    const __bf16* __restrict__ qrep,    // [QN][LQ][HH]
    const __bf16* __restrict__ flow_t,  // [QN][HH][KPAD]
    const int*  __restrict__ tags,      // [QN]
    float*       __restrict__ out)      // [QN][LC][HH]
{
    __shared__ float  sc[4][16][48];    // raw scores, wave-private rows
    __shared__ __bf16 at[4][16][KPAD];  // attn weights, K padded 48 -> 64

    const int q     = blockIdx.x >> 4;  // / (LC/64)
    const int chunk = blockIdx.x & 15;
    const int wave  = threadIdx.x >> 5;
    const int lane  = threadIdx.x & 31;
    const int m  = lane & 15;
    const int hf = lane >> 4;

    const int rowBase = chunk * 64 + wave * 16;
    const __bf16* aRow = crep + ((size_t)tags[q] * LC + rowBase + m) * HH + hf * 16;
    const __bf16* qr   = qrep + (size_t)q * LQ * HH + (size_t)m * HH + hf * 16;

    // ---- scores: [16 rows] x [48 cols], K = HH in steps of 32 ----
    v8f s0 = {}, s1 = {}, s2 = {};
    for (int kb = 0; kb < HH; kb += 32) {
        v16bf a  = load16v(aRow + kb);
        v16bf b0 = load16v(qr + kb);                       // cols  0..15
        v16bf b1 = load16v(qr + (size_t)16 * HH + kb);     // cols 16..31
        v16bf b2 = load16v(qr + (size_t)32 * HH + kb);     // cols 32..47
        s0 = wmma_bf16(a, b0, s0);
        s1 = wmma_bf16(a, b1, s1);
        s2 = wmma_bf16(a, b2, s2);
    }

    // scatter scores to LDS (lane: col = m + 16*t, rows r + 8*hf)
#pragma unroll
    for (int r = 0; r < 8; ++r) {
        int mr = r + 8 * hf;
        sc[wave][mr][m]      = s0[r];
        sc[wave][mr][m + 16] = s1[r];
        sc[wave][mr][m + 32] = s2[r];
    }
    __syncthreads();

    // ---- softmax over 48, one thread per row (64 rows total) ----
    if (threadIdx.x < 64) {
        int w = threadIdx.x >> 4, r = threadIdx.x & 15;
        float mx = -1e30f;
#pragma unroll
        for (int j = 0; j < 48; ++j) mx = fmaxf(mx, sc[w][r][j]);
        float e[48];
        float sum = 0.f;
#pragma unroll
        for (int j = 0; j < 48; ++j) { e[j] = __expf(sc[w][r][j] - mx); sum += e[j]; }
        float inv = 1.0f / sum;
#pragma unroll
        for (int j = 0; j < 48; ++j) at[w][r][j] = (__bf16)(e[j] * inv);
#pragma unroll
        for (int j = 48; j < KPAD; ++j) at[w][r][j] = (__bf16)0.f;
    }
    __syncthreads();

    // ---- out = attn[16x64] @ flow_t^T, f32 out; B frags contiguous in flow_t ----
    v16bf a0 = load16v(&at[wave][m][hf * 16]);        // k =  0..31 half
    v16bf a1 = load16v(&at[wave][m][32 + hf * 16]);   // k = 32..63 half (pad zeros)
    const __bf16* ft = flow_t + (size_t)q * HH * KPAD;
    float* orow = out + ((size_t)q * LC + rowBase) * HH;

    for (int n0 = 0; n0 < HH; n0 += 16) {
        const __bf16* fb = ft + (size_t)(n0 + m) * KPAD + hf * 16;
        v8f o = {};
        o = wmma_bf16(a0, load16v(fb), o);        // k block 0..31
        o = wmma_bf16(a1, load16v(fb + 32), o);   // k block 32..63
#pragma unroll
        for (int r = 0; r < 8; ++r)
            orow[(size_t)(r + 8 * hf) * HH + n0 + m] = o[r];
    }
}

// ---------------------------------------------------------------------------
// Launch
// ---------------------------------------------------------------------------
extern "C" void kernel_launch(void* const* d_in, const int* in_sizes, int n_in,
                              void* d_out, int out_size, void* d_ws, size_t ws_size,
                              hipStream_t stream) {
    const float* contexts  = (const float*)d_in[0];   // [8][1024][768]
    const float* questions = (const float*)d_in[1];   // [64][48][768]
    const int*   tags      = (const int*)  d_in[2];   // [64]
    const float* W1        = (const float*)d_in[3];   // [768][1536]
    const float* W2        = (const float*)d_in[4];   // [1536][1536]
    const float* v         = (const float*)d_in[5];   // [1536]
    float* out = (float*)d_out;                       // [64][1024][1536]

    // workspace layout (bf16 buffers), 256B aligned
    const size_t N_FLOW  = (size_t)QN * LQ * HH;      //  4,718,592
    const size_t N_FLOWT = (size_t)QN * HH * KPAD;    //  6,291,456
    const size_t N_CTX   = (size_t)BN_ * LC * DC;     //  6,291,456
    const size_t N_W1    = (size_t)DC * HH;           //  1,179,648
    const size_t N_W2    = (size_t)HH * HH;           //  2,359,296
    const size_t N_CREP  = (size_t)BN_ * LC * HH;     // 12,582,912
    const size_t N_QREP  = (size_t)QN * LQ * HH;      //  4,718,592

    char* ws = (char*)d_ws;
    size_t off = 0;
    auto take = [&](size_t elems) {
        void* p = ws + off;
        off = (off + elems * sizeof(__bf16) + 255) & ~(size_t)255;
        return (__bf16*)p;
    };
    __bf16* flow_b  = take(N_FLOW);
    __bf16* flowt_b = take(N_FLOWT);
    __bf16* ctx_b   = take(N_CTX);
    __bf16* w1_b    = take(N_W1);
    __bf16* w2_b    = take(N_W2);
    __bf16* crep_b  = take(N_CREP);
    __bf16* qrep_b  = take(N_QREP);

    // 1) conversions to bf16 (vectorized x4)
    f32_to_bf16_v4<<<(int)((N_CTX / 4 + 255) / 256), 256, 0, stream>>>(contexts, ctx_b, (int)(N_CTX / 4));
    f32_to_bf16_v4<<<(int)((N_W1  / 4 + 255) / 256), 256, 0, stream>>>(W1, w1_b, (int)(N_W1 / 4));
    f32_to_bf16_v4<<<(int)((N_W2  / 4 + 255) / 256), 256, 0, stream>>>(W2, w2_b, (int)(N_W2 / 4));

    // 2) question flow (group prefix-mean scan) -> bf16 flow + transposed flow
    flow_kernel<<<(LQ * DQ + 255) / 256, 256, 0, stream>>>(questions, tags, flow_b, flowt_b);

    // 3) c_rep = relu(ctx @ W1):  [8192 x 768] @ [768 x 1536]
    gemm_relu_bf16<BN_ * LC, HH, DC><<<dim3((BN_ * LC) / 128, HH / 64), 256, 0, stream>>>(
        ctx_b, w1_b, crep_b, nullptr);

    // 4) q_rep = v * relu(flow @ W2):  [3072 x 1536] @ [1536 x 1536]
    gemm_relu_bf16<QN * LQ, HH, HH><<<dim3((QN * LQ) / 128, HH / 64), 256, 0, stream>>>(
        flow_b, w2_b, qrep_b, v);

    // 5) fused scores -> softmax -> out, per (question, 64-row Lc chunk)
    attn_kernel<<<QN * (LC / 64), 128, 0, stream>>>(crep_b, qrep_b, flowt_b, tags, out);
}
